// EnhancedCrossAttentionFusion_81990925680992
// MI455X (gfx1250) — compile-verified, matrix operand order checked by reading, not run on hardware
//
#include <hip/hip_runtime.h>
#include <hip/hip_bf16.h>
#include <cstdint>

// ---------------------------------------------------------------------------
// EnhancedCrossAttentionFusion for MI455X (gfx1250, wave32, WMMA)
// B=4096, H=1024, NH=16, HD=64, EF=4
// All large GEMMs via v_wmma_f32_16x16x32_f16 with double-buffered LDS
// (async global->LDS DMA when available), tiny S=4 attention via
// wave-per-head VALU, fused LN / gating epilogues.
// ---------------------------------------------------------------------------

typedef _Float16 v16h __attribute__((ext_vector_type(16)));
typedef float    v8f  __attribute__((ext_vector_type(8)));
typedef int      v4i_ __attribute__((ext_vector_type(4)));

#define BM 128
#define BN 128
#define BK 32
#define LDSW 40   // 40 halves = 80B row stride: 16B aligned, conflict-free

// CDNA5 async global->LDS DMA (ASYNCcnt-tracked), if the toolchain has it.
#if defined(__has_builtin)
#if __has_builtin(__builtin_amdgcn_global_load_async_to_lds_b128) && \
    __has_builtin(__builtin_amdgcn_s_wait_asynccnt)
#define USE_ASYNC_LDS 1
#endif
#endif
#ifndef USE_ASYNC_LDS
#define USE_ASYNC_LDS 0
#endif

#if USE_ASYNC_LDS
// builtin signature: (v4i addrspace(1)* gsrc, v4i addrspace(3)* ldst, imm off, imm cpol)
typedef __attribute__((address_space(1))) v4i_ glb_v4i;
typedef __attribute__((address_space(3))) v4i_ lds_v4i;
__device__ __forceinline__ void glb_to_lds_b128(const _Float16* g, _Float16* l) {
  v4i_* gp = (v4i_*)(void*)const_cast<_Float16*>(g);
  v4i_* lp = (v4i_*)(void*)l;
  __builtin_amdgcn_global_load_async_to_lds_b128(
      (glb_v4i*)gp, (lds_v4i*)lp, 0, 0);
}
#endif

// ============================ WMMA GEMM ====================================
// C[M,N] = act( A[M,K] * W[N,K]^T + bias[N] ),   A,W f16; C f32 and/or f16.
// W stored [N,K] row-major == the K x N B-matrix WMMA wants (no transpose).
template <int ACT>  // 0=none, 1=relu, 2=sigmoid
__global__ __launch_bounds__(256) void gemm_f16_kernel(
    const _Float16* __restrict__ A, int lda,
    const _Float16* __restrict__ W, int ldw,
    const float* __restrict__ bias,
    float* __restrict__ Cf, _Float16* __restrict__ Ch, int ldc,
    int K)
{
  __shared__ _Float16 As[2][BM * LDSW];
  __shared__ _Float16 Bs[2][BN * LDSW];

  const int tid  = threadIdx.x;
  const int lane = tid & 31;
  const int wid  = tid >> 5;
  const int wm   = wid >> 1;      // 4 wave-rows  (32 M each)
  const int wn   = wid & 1;       // 2 wave-cols  (64 N each)
  const int m0   = blockIdx.y * BM;
  const int n0   = blockIdx.x * BN;
  const int hi   = lane >> 4;     // half-wave select
  const int ml   = lane & 15;

  v8f acc[2][4];
#pragma unroll
  for (int i = 0; i < 2; ++i)
#pragma unroll
    for (int j = 0; j < 4; ++j) acc[i][j] = (v8f){0, 0, 0, 0, 0, 0, 0, 0};

  // cooperative tile load mapping: 64 rows x 4 chunks-of-8-halves per pass
  const int lr = tid >> 2;        // 0..63
  const int lc = (tid & 3) * 8;   // 0,8,16,24 (16B chunk)

  // hoisted global row pointers (k advances by immediate only)
  const _Float16* ga0 = A + (size_t)(m0 + lr) * lda + lc;
  const _Float16* ga1 = ga0 + (size_t)64 * lda;
  const _Float16* gb0 = W + (size_t)(n0 + lr) * ldw + lc;
  const _Float16* gb1 = gb0 + (size_t)64 * ldw;

  // hoisted LDS staging pointers (both buffers)
  _Float16* sa0[2] = {&As[0][lr * LDSW + lc], &As[1][lr * LDSW + lc]};
  _Float16* sa1[2] = {&As[0][(lr + 64) * LDSW + lc], &As[1][(lr + 64) * LDSW + lc]};
  _Float16* sb0[2] = {&Bs[0][lr * LDSW + lc], &Bs[1][lr * LDSW + lc]};
  _Float16* sb1[2] = {&Bs[0][(lr + 64) * LDSW + lc], &Bs[1][(lr + 64) * LDSW + lc]};

  // hoisted fragment offsets (16-bit A layout: lanes 0-15 K=0..7|16..23,
  // lanes 16-31 K=8..15|24..31 -> two b128 LDS loads per fragment)
  int aoff[2], boff[4];
#pragma unroll
  for (int i = 0; i < 2; ++i) aoff[i] = (wm * 32 + i * 16 + ml) * LDSW + hi * 8;
#pragma unroll
  for (int j = 0; j < 4; ++j) boff[j] = (wn * 64 + j * 16 + ml) * LDSW + hi * 8;

  union Frag { uint4 u[2]; v16h v; };
  auto compute_tile = [&](int b_) {
    Frag a[2], b[4];
#pragma unroll
    for (int i = 0; i < 2; ++i) {
      a[i].u[0] = *(const uint4*)(&As[b_][aoff[i]]);
      a[i].u[1] = *(const uint4*)(&As[b_][aoff[i] + 16]);
    }
#pragma unroll
    for (int j = 0; j < 4; ++j) {
      b[j].u[0] = *(const uint4*)(&Bs[b_][boff[j]]);
      b[j].u[1] = *(const uint4*)(&Bs[b_][boff[j] + 16]);
    }
#pragma unroll
    for (int i = 0; i < 2; ++i)
#pragma unroll
      for (int j = 0; j < 4; ++j)
        acc[i][j] = __builtin_amdgcn_wmma_f32_16x16x32_f16(
            false, a[i].v, false, b[j].v, (short)0, acc[i][j], false, false);
  };

  // -------- prologue: tile 0 into buffer 0 --------
#if USE_ASYNC_LDS
  glb_to_lds_b128(ga0, sa0[0]);
  glb_to_lds_b128(ga1, sa1[0]);
  glb_to_lds_b128(gb0, sb0[0]);
  glb_to_lds_b128(gb1, sb1[0]);
  __builtin_amdgcn_s_wait_asynccnt(0);
#else
  {
    uint4 ra0 = *(const uint4*)ga0, ra1 = *(const uint4*)ga1;
    uint4 rb0 = *(const uint4*)gb0, rb1 = *(const uint4*)gb1;
    *(uint4*)sa0[0] = ra0; *(uint4*)sa1[0] = ra1;
    *(uint4*)sb0[0] = rb0; *(uint4*)sb1[0] = rb1;
  }
#endif
  __syncthreads();

  // -------- steady state: prefetch kb+BK into buf^1 while computing buf ----
  int buf = 0;
  for (int kb = 0; kb < K - BK; kb += BK) {
    const int kn = kb + BK;
#if USE_ASYNC_LDS
    glb_to_lds_b128(ga0 + kn, sa0[buf ^ 1]);
    glb_to_lds_b128(ga1 + kn, sa1[buf ^ 1]);
    glb_to_lds_b128(gb0 + kn, sb0[buf ^ 1]);
    glb_to_lds_b128(gb1 + kn, sb1[buf ^ 1]);
    compute_tile(buf);
    __builtin_amdgcn_s_wait_asynccnt(0);
#else
    uint4 ra0 = *(const uint4*)(ga0 + kn), ra1 = *(const uint4*)(ga1 + kn);
    uint4 rb0 = *(const uint4*)(gb0 + kn), rb1 = *(const uint4*)(gb1 + kn);
    compute_tile(buf);   // loads in flight under the WMMAs
    *(uint4*)sa0[buf ^ 1] = ra0; *(uint4*)sa1[buf ^ 1] = ra1;
    *(uint4*)sb0[buf ^ 1] = rb0; *(uint4*)sb1[buf ^ 1] = rb1;
#endif
    __syncthreads();
    buf ^= 1;
  }
  compute_tile(buf);   // last tile, no prefetch

  // -------- epilogue: D layout n = lane&15 ; vgpr r -> m = r + 8*(lane>>4) --
#pragma unroll
  for (int i = 0; i < 2; ++i) {
#pragma unroll
    for (int j = 0; j < 4; ++j) {
      const int n  = n0 + wn * 64 + j * 16 + ml;
      const float bv = bias ? bias[n] : 0.f;
#pragma unroll
      for (int r = 0; r < 8; ++r) {
        const int m = m0 + wm * 32 + i * 16 + r + hi * 8;
        float v = acc[i][j][r] + bv;
        if (ACT == 1) v = fmaxf(v, 0.f);
        if (ACT == 2) v = 1.f / (1.f + __expf(-v));
        const size_t off = (size_t)m * ldc + n;
        if (Cf) Cf[off] = v;
        if (Ch) Ch[off] = (_Float16)v;
      }
    }
  }
}

static void launch_gemm(int act, const _Float16* A, int lda,
                        const _Float16* W, int ldw, const float* bias,
                        float* Cf, _Float16* Ch, int ldc,
                        int M, int N, int K, hipStream_t s)
{
  dim3 g(N / BN, M / BM), b(256);
  if (act == 0)      gemm_f16_kernel<0><<<g, b, 0, s>>>(A, lda, W, ldw, bias, Cf, Ch, ldc, K);
  else if (act == 1) gemm_f16_kernel<1><<<g, b, 0, s>>>(A, lda, W, ldw, bias, Cf, Ch, ldc, K);
  else               gemm_f16_kernel<2><<<g, b, 0, s>>>(A, lda, W, ldw, bias, Cf, Ch, ldc, K);
}

// ====================== tiny S=4 attention (per-head) ======================
// qkv f32 [16384,3072] rows (b*4+s), Q|K|V segments; out f16 [16384,1024].
__global__ __launch_bounds__(256) void attn4_kernel(
    const float* __restrict__ qkv, _Float16* __restrict__ out, int total)
{
  const int wid  = threadIdx.x >> 5;
  const int lane = threadIdx.x & 31;
  const int g    = blockIdx.x * 8 + wid;
  if (g >= total) return;
  const int b = g >> 4, h = g & 15;
  const size_t row0 = (size_t)b * 4;
  const float* base = qkv + row0 * 3072 + h * 64 + lane * 2;

  float2 q[4], k[4], v[4];
#pragma unroll
  for (int s = 0; s < 4; ++s) {
    q[s] = *(const float2*)(base + (size_t)s * 3072);
    k[s] = *(const float2*)(base + (size_t)s * 3072 + 1024);
    v[s] = *(const float2*)(base + (size_t)s * 3072 + 2048);
  }
  float sc[4][4];
#pragma unroll
  for (int i = 0; i < 4; ++i)
#pragma unroll
    for (int j = 0; j < 4; ++j) {
      float p = q[i].x * k[j].x + q[i].y * k[j].y;
#pragma unroll
      for (int o = 16; o > 0; o >>= 1) p += __shfl_xor(p, o, 32);
      sc[i][j] = p * 0.125f;  // 1/sqrt(64)
    }
#pragma unroll
  for (int i = 0; i < 4; ++i) {
    const float mx = fmaxf(fmaxf(sc[i][0], sc[i][1]), fmaxf(sc[i][2], sc[i][3]));
    const float e0 = __expf(sc[i][0] - mx), e1 = __expf(sc[i][1] - mx);
    const float e2 = __expf(sc[i][2] - mx), e3 = __expf(sc[i][3] - mx);
    const float inv = 1.f / (e0 + e1 + e2 + e3);
    const float ox = (e0 * v[0].x + e1 * v[1].x + e2 * v[2].x + e3 * v[3].x) * inv;
    const float oy = (e0 * v[0].y + e1 * v[1].y + e2 * v[2].y + e3 * v[3].y) * inv;
    union { _Float16 h2[2]; unsigned u; } pk;
    pk.h2[0] = (_Float16)ox; pk.h2[1] = (_Float16)oy;
    *(unsigned*)(out + (row0 + i) * 1024 + h * 64 + lane * 2) = pk.u;
  }
}

// ======================= LayerNorm over H=1024 =============================
template <bool RES>
__global__ __launch_bounds__(256) void layernorm1024_kernel(
    const float* __restrict__ X, const float* __restrict__ R,
    const float* __restrict__ gamma, const float* __restrict__ beta,
    float* __restrict__ Yf, _Float16* __restrict__ Yh)
{
  const int row = blockIdx.x, tid = threadIdx.x;
  const size_t base = (size_t)row * 1024;
  float v[4], s = 0.f, s2 = 0.f;
#pragma unroll
  for (int i = 0; i < 4; ++i) {
    const int c = tid + i * 256;
    float x = X[base + c];
    if (RES) x += R[base + c];
    v[i] = x; s += x; s2 += x * x;
  }
#pragma unroll
  for (int o = 16; o > 0; o >>= 1) { s += __shfl_xor(s, o, 32); s2 += __shfl_xor(s2, o, 32); }
  __shared__ float sh[16];
  const int wid = tid >> 5, lane = tid & 31;
  if (lane == 0) { sh[wid] = s; sh[8 + wid] = s2; }
  __syncthreads();
  float ts = 0.f, ts2 = 0.f;
#pragma unroll
  for (int i = 0; i < 8; ++i) { ts += sh[i]; ts2 += sh[8 + i]; }
  const float mean = ts * (1.f / 1024.f);
  const float var  = ts2 * (1.f / 1024.f) - mean * mean;
  const float inv  = rsqrtf(var + 1e-5f);
#pragma unroll
  for (int i = 0; i < 4; ++i) {
    const int c = tid + i * 256;
    const float y = (v[i] - mean) * inv * gamma[c] + beta[c];
    if (Yf) Yf[base + c] = y;
    if (Yh) Yh[base + c] = (_Float16)y;
  }
}

// LN3 + learned residual mix -> final output
__global__ __launch_bounds__(256) void ln3_final_kernel(
    const float* __restrict__ X, const float* __restrict__ gamma,
    const float* __restrict__ beta, const float* __restrict__ resw,
    const float* __restrict__ tf, const float* __restrict__ sf,
    float* __restrict__ out)
{
  const int row = blockIdx.x, tid = threadIdx.x;
  const size_t base = (size_t)row * 1024;
  float v[4], s = 0.f, s2 = 0.f;
#pragma unroll
  for (int i = 0; i < 4; ++i) {
    const int c = tid + i * 256;
    const float x = X[base + c];
    v[i] = x; s += x; s2 += x * x;
  }
#pragma unroll
  for (int o = 16; o > 0; o >>= 1) { s += __shfl_xor(s, o, 32); s2 += __shfl_xor(s2, o, 32); }
  __shared__ float sh[16];
  const int wid = tid >> 5, lane = tid & 31;
  if (lane == 0) { sh[wid] = s; sh[8 + wid] = s2; }
  __syncthreads();
  float ts = 0.f, ts2 = 0.f;
#pragma unroll
  for (int i = 0; i < 8; ++i) { ts += sh[i]; ts2 += sh[8 + i]; }
  const float mean = ts * (1.f / 1024.f);
  const float var  = ts2 * (1.f / 1024.f) - mean * mean;
  const float inv  = rsqrtf(var + 1e-5f);
  const float w    = resw[0];
#pragma unroll
  for (int i = 0; i < 4; ++i) {
    const int c = tid + i * 256;
    const float y = (v[i] - mean) * inv * gamma[c] + beta[c];
    const float orig = 0.5f * (tf[base + c] + sf[base + c]);
    out[base + c] = w * y + (1.f - w) * orig;
  }
}

// ======================= small elementwise kernels =========================
__global__ void f32_to_f16_kernel(const float* __restrict__ X,
                                  _Float16* __restrict__ Y, size_t n) {
  const size_t i = (size_t)blockIdx.x * 256 + threadIdx.x;
  if (i < n) Y[i] = (_Float16)X[i];
}

__global__ void addvec_kernel(const float* __restrict__ a,
                              const float* __restrict__ b,
                              float* __restrict__ y, int n) {
  const int i = blockIdx.x * 256 + threadIdx.x;
  if (i < n) y[i] = a[i] + b[i];
}

// mean over EF=4 rows: X [B*4,1024] -> Y [B,1024]
__global__ void mean4_kernel(const float* __restrict__ X,
                             float* __restrict__ Y, int n) {
  const int i = blockIdx.x * 256 + threadIdx.x;
  if (i >= n) return;
  const int b = i >> 10, c = i & 1023;
  const float* p = X + (size_t)b * 4096 + c;
  Y[i] = 0.25f * (p[0] + p[1024] + p[2048] + p[3072]);
}

// concat(mean4(T), mean4(S)) -> f16 [B,2048]
__global__ void meanconcat_kernel(const float* __restrict__ T,
                                  const float* __restrict__ S,
                                  _Float16* __restrict__ Y, int n) {
  const int i = blockIdx.x * 256 + threadIdx.x;
  if (i >= n) return;
  const int b = i >> 11, c = i & 2047;
  const float* src = (c < 1024) ? T : S;
  const int cc = c & 1023;
  const float* p = src + (size_t)b * 4096 + cc;
  Y[i] = (_Float16)(0.25f * (p[0] + p[1024] + p[2048] + p[3072]));
}

// concat of two f32 [B,1024] -> f16 [B,2048]
__global__ void concat2h_kernel(const float* __restrict__ A,
                                const float* __restrict__ B,
                                _Float16* __restrict__ Y, int n) {
  const int i = blockIdx.x * 256 + threadIdx.x;
  if (i >= n) return;
  const int b = i >> 11, c = i & 2047;
  Y[i] = (_Float16)((c < 1024) ? A[(size_t)b * 1024 + c]
                               : B[(size_t)b * 1024 + (c - 1024)]);
}

// broadcast joint [B,1024] -> f16 [B*4,1024]
__global__ void bcast4h_kernel(const float* __restrict__ J,
                               _Float16* __restrict__ Y, int n) {
  const int i = blockIdx.x * 256 + threadIdx.x;
  if (i >= n) return;
  const int r = i >> 10, c = i & 1023;
  Y[i] = (_Float16)J[(size_t)(r >> 2) * 1024 + c];
}

// fused = concat(gt*mts, gs*mst, gt*mtj, gs*msj, gj*mjt, gj*mjs) -> f16 [B,6144]
__global__ void fused_build_kernel(
    const float* __restrict__ gt, const float* __restrict__ gs,
    const float* __restrict__ gj,
    const float* __restrict__ mts, const float* __restrict__ mst,
    const float* __restrict__ mtj, const float* __restrict__ msj,
    const float* __restrict__ mjt, const float* __restrict__ mjs,
    _Float16* __restrict__ Y, int n)
{
  const int i = blockIdx.x * 256 + threadIdx.x;
  if (i >= n) return;
  const int b = i / 6144, c = i % 6144;
  const int seg = c >> 10, col = c & 1023;
  const size_t o = (size_t)b * 1024 + col;
  const float* gsel[6] = {gt, gs, gt, gs, gj, gj};
  const float* msel[6] = {mts, mst, mtj, msj, mjt, mjs};
  Y[i] = (_Float16)(gsel[seg][o] * msel[seg][o]);
}

// ============================ host orchestration ===========================
static inline dim3 ewg(size_t n) { return dim3((unsigned)((n + 255) / 256)); }

static void run_mha(const _Float16* q_h, const _Float16* kv_h, bool self_attn,
                    const _Float16* in_w_h, const float* in_b,
                    const _Float16* out_w_h, const float* out_b,
                    float* qkv_f, _Float16* attn_h, float* proj_f,
                    hipStream_t s)
{
  const int M = 16384;
  if (self_attn) {
    launch_gemm(0, q_h, 1024, in_w_h, 1024, in_b, qkv_f, nullptr, 3072,
                M, 3072, 1024, s);
  } else {
    launch_gemm(0, q_h, 1024, in_w_h, 1024, in_b, qkv_f, nullptr, 3072,
                M, 1024, 1024, s);
    launch_gemm(0, kv_h, 1024, in_w_h + (size_t)1024 * 1024, 1024,
                in_b + 1024, qkv_f + 1024, nullptr, 3072, M, 2048, 1024, s);
  }
  attn4_kernel<<<dim3(65536 / 8), 256, 0, s>>>(qkv_f, attn_h, 65536);
  launch_gemm(0, attn_h, 1024, out_w_h, 1024, out_b, proj_f, nullptr, 1024,
              M, 1024, 1024, s);
}

extern "C" void kernel_launch(void* const* d_in, const int* in_sizes, int n_in,
                              void* d_out, int out_size, void* d_ws, size_t ws_size,
                              hipStream_t stream)
{
  (void)in_sizes; (void)n_in; (void)out_size; (void)ws_size;
  const int B = 4096, H = 1024;
  const int MBE = B * 4;  // 16384 rows for [B,EF,H]

  // ---- inputs (setup_inputs order) ----
  const float* t_feat   = (const float*)d_in[0];
  const float* s_feat   = (const float*)d_in[1];
  const float* te_w     = (const float*)d_in[2];
  const float* te_b     = (const float*)d_in[3];
  const float* se_w     = (const float*)d_in[4];
  const float* se_b     = (const float*)d_in[5];
  const float* pos_enc  = (const float*)d_in[6];
  const float* in_w[5]  = {(const float*)d_in[7],  (const float*)d_in[11],
                           (const float*)d_in[15], (const float*)d_in[19],
                           (const float*)d_in[23]};
  const float* in_b[5]  = {(const float*)d_in[8],  (const float*)d_in[12],
                           (const float*)d_in[16], (const float*)d_in[20],
                           (const float*)d_in[24]};
  const float* out_w[5] = {(const float*)d_in[9],  (const float*)d_in[13],
                           (const float*)d_in[17], (const float*)d_in[21],
                           (const float*)d_in[25]};
  const float* out_b[5] = {(const float*)d_in[10], (const float*)d_in[14],
                           (const float*)d_in[18], (const float*)d_in[22],
                           (const float*)d_in[26]};
  const float* jp_w = (const float*)d_in[27]; const float* jp_b = (const float*)d_in[28];
  const float* gt_w = (const float*)d_in[29]; const float* gt_b = (const float*)d_in[30];
  const float* gs_w = (const float*)d_in[31]; const float* gs_b = (const float*)d_in[32];
  const float* gj_w = (const float*)d_in[33]; const float* gj_b = (const float*)d_in[34];
  const float* op1_w = (const float*)d_in[35]; const float* op1_b = (const float*)d_in[36];
  const float* op2_w = (const float*)d_in[37]; const float* op2_b = (const float*)d_in[38];
  const float* ln1_g = (const float*)d_in[39]; const float* ln1_b = (const float*)d_in[40];
  const float* ln2_g = (const float*)d_in[41]; const float* ln2_b = (const float*)d_in[42];
  const float* ln3_g = (const float*)d_in[43]; const float* ln3_b = (const float*)d_in[44];
  const float* res_w = (const float*)d_in[45];
  float* out = (float*)d_out;

  // ---- workspace bump allocator ----
  size_t off = 0;
  char* base = (char*)d_ws;
  auto alloc = [&](size_t elems, size_t esz) -> void* {
    off = (off + 255) & ~(size_t)255;
    void* p = base + off;
    off += elems * esz;
    return p;
  };
  auto ah = [&](size_t e) { return (_Float16*)alloc(e, 2); };
  auto af = [&](size_t e) { return (float*)alloc(e, 4); };

  // f16 weights
  _Float16* te_w_h  = ah((size_t)4096 * 1024);
  _Float16* se_w_h  = ah((size_t)4096 * 1024);
  _Float16* in_w_h[5], * out_w_h[5];
  for (int i = 0; i < 5; ++i) { in_w_h[i] = ah((size_t)3072 * 1024); out_w_h[i] = ah((size_t)1024 * 1024); }
  _Float16* jp_w_h  = ah((size_t)1024 * 2048);
  _Float16* gt_w_h  = ah((size_t)1024 * 2048);
  _Float16* gs_w_h  = ah((size_t)1024 * 2048);
  _Float16* gj_w_h  = ah((size_t)1024 * 2048);
  _Float16* op1_w_h = ah((size_t)2048 * 6144);
  _Float16* op2_w_h = ah((size_t)1024 * 2048);
  // f16 activations
  _Float16* xt_h   = ah((size_t)B * H);
  _Float16* xs_h   = ah((size_t)B * H);
  _Float16* texp_h = ah((size_t)MBE * H);
  _Float16* sexp_h = ah((size_t)MBE * H);
  _Float16* tenh_h = ah((size_t)MBE * H);
  _Float16* senh_h = ah((size_t)MBE * H);
  _Float16* jexp_h = ah((size_t)MBE * H);
  _Float16* attn_h = ah((size_t)MBE * H);
  _Float16* cat_h  = ah((size_t)B * 2048);
  _Float16* fused_h= ah((size_t)B * 6144);
  _Float16* h1_h   = ah((size_t)B * 2048);
  // f32 buffers
  float* posb_t = af(4096);
  float* posb_s = af(4096);
  float* texp_f = af((size_t)MBE * H);
  float* sexp_f = af((size_t)MBE * H);
  float* qkv_f  = af((size_t)MBE * 3072);
  float* proj_f = af((size_t)MBE * H);
  float* tenh_f = af((size_t)MBE * H);
  float* senh_f = af((size_t)MBE * H);
  float* joint_f= af((size_t)B * H);
  float* m_ts = af((size_t)B * H); float* m_st = af((size_t)B * H);
  float* m_tj = af((size_t)B * H); float* m_sj = af((size_t)B * H);
  float* m_jt = af((size_t)B * H); float* m_js = af((size_t)B * H);
  float* gate_t = af((size_t)B * H); float* gate_s = af((size_t)B * H);
  float* gate_j = af((size_t)B * H);
  float* ln3in_f = af((size_t)B * H);

  auto cvt = [&](const float* src, _Float16* dst, size_t n) {
    f32_to_f16_kernel<<<ewg(n), 256, 0, stream>>>(src, dst, n);
  };

  // ---- 0) convert weights + inputs to f16 ----
  cvt(te_w, te_w_h, (size_t)4096 * 1024);
  cvt(se_w, se_w_h, (size_t)4096 * 1024);
  for (int i = 0; i < 5; ++i) {
    cvt(in_w[i],  in_w_h[i],  (size_t)3072 * 1024);
    cvt(out_w[i], out_w_h[i], (size_t)1024 * 1024);
  }
  cvt(jp_w, jp_w_h, (size_t)1024 * 2048);
  cvt(gt_w, gt_w_h, (size_t)1024 * 2048);
  cvt(gs_w, gs_w_h, (size_t)1024 * 2048);
  cvt(gj_w, gj_w_h, (size_t)1024 * 2048);
  cvt(op1_w, op1_w_h, (size_t)2048 * 6144);
  cvt(op2_w, op2_w_h, (size_t)1024 * 2048);
  cvt(t_feat, xt_h, (size_t)B * H);
  cvt(s_feat, xs_h, (size_t)B * H);

  // combined bias = expand-bias + flattened pos_enc (column n == e*H+h)
  addvec_kernel<<<ewg(4096), 256, 0, stream>>>(te_b, pos_enc, posb_t, 4096);
  addvec_kernel<<<ewg(4096), 256, 0, stream>>>(se_b, pos_enc, posb_s, 4096);

  // ---- 1) expansions: [B,H] x [4H,H]^T -> [B,4H] == [B*4,H] ----
  launch_gemm(0, xt_h, 1024, te_w_h, 1024, posb_t, texp_f, texp_h, 4096,
              B, 4096, 1024, stream);
  launch_gemm(0, xs_h, 1024, se_w_h, 1024, posb_s, sexp_f, sexp_h, 4096,
              B, 4096, 1024, stream);

  // ---- 2) self-attention + residual LN ----
  run_mha(texp_h, texp_h, true, in_w_h[0], in_b[0], out_w_h[0], out_b[0],
          qkv_f, attn_h, proj_f, stream);
  layernorm1024_kernel<true><<<MBE, 256, 0, stream>>>(
      proj_f, texp_f, ln1_g, ln1_b, tenh_f, tenh_h);
  run_mha(sexp_h, sexp_h, true, in_w_h[1], in_b[1], out_w_h[1], out_b[1],
          qkv_f, attn_h, proj_f, stream);
  layernorm1024_kernel<true><<<MBE, 256, 0, stream>>>(
      proj_f, sexp_f, ln2_g, ln2_b, senh_f, senh_h);

  // ---- 3) joint features ----
  meanconcat_kernel<<<ewg((size_t)B * 2048), 256, 0, stream>>>(
      tenh_f, senh_f, cat_h, B * 2048);
  launch_gemm(0, cat_h, 2048, jp_w_h, 2048, jp_b, joint_f, nullptr, 1024,
              B, 1024, 2048, stream);
  bcast4h_kernel<<<ewg((size_t)MBE * H), 256, 0, stream>>>(joint_f, jexp_h, MBE * H);

  // ---- 4) six cross-attentions -> per-branch means ----
  run_mha(tenh_h, senh_h, false, in_w_h[2], in_b[2], out_w_h[2], out_b[2],
          qkv_f, attn_h, proj_f, stream);
  mean4_kernel<<<ewg((size_t)B * H), 256, 0, stream>>>(proj_f, m_ts, B * H);
  run_mha(senh_h, tenh_h, false, in_w_h[3], in_b[3], out_w_h[3], out_b[3],
          qkv_f, attn_h, proj_f, stream);
  mean4_kernel<<<ewg((size_t)B * H), 256, 0, stream>>>(proj_f, m_st, B * H);
  run_mha(tenh_h, jexp_h, false, in_w_h[4], in_b[4], out_w_h[4], out_b[4],
          qkv_f, attn_h, proj_f, stream);
  mean4_kernel<<<ewg((size_t)B * H), 256, 0, stream>>>(proj_f, m_tj, B * H);
  run_mha(senh_h, jexp_h, false, in_w_h[4], in_b[4], out_w_h[4], out_b[4],
          qkv_f, attn_h, proj_f, stream);
  mean4_kernel<<<ewg((size_t)B * H), 256, 0, stream>>>(proj_f, m_sj, B * H);
  run_mha(jexp_h, tenh_h, false, in_w_h[4], in_b[4], out_w_h[4], out_b[4],
          qkv_f, attn_h, proj_f, stream);
  mean4_kernel<<<ewg((size_t)B * H), 256, 0, stream>>>(proj_f, m_jt, B * H);
  run_mha(jexp_h, senh_h, false, in_w_h[4], in_b[4], out_w_h[4], out_b[4],
          qkv_f, attn_h, proj_f, stream);
  mean4_kernel<<<ewg((size_t)B * H), 256, 0, stream>>>(proj_f, m_js, B * H);

  // ---- 5) gates (sigmoid fused into GEMM epilogue) ----
  concat2h_kernel<<<ewg((size_t)B * 2048), 256, 0, stream>>>(m_ts, m_tj, cat_h, B * 2048);
  launch_gemm(2, cat_h, 2048, gt_w_h, 2048, gt_b, gate_t, nullptr, 1024, B, 1024, 2048, stream);
  concat2h_kernel<<<ewg((size_t)B * 2048), 256, 0, stream>>>(m_st, m_sj, cat_h, B * 2048);
  launch_gemm(2, cat_h, 2048, gs_w_h, 2048, gs_b, gate_s, nullptr, 1024, B, 1024, 2048, stream);
  concat2h_kernel<<<ewg((size_t)B * 2048), 256, 0, stream>>>(m_jt, m_js, cat_h, B * 2048);
  launch_gemm(2, cat_h, 2048, gj_w_h, 2048, gj_b, gate_j, nullptr, 1024, B, 1024, 2048, stream);

  // ---- 6) fuse + output projection ----
  fused_build_kernel<<<ewg((size_t)B * 6144), 256, 0, stream>>>(
      gate_t, gate_s, gate_j, m_ts, m_st, m_tj, m_sj, m_jt, m_js,
      fused_h, B * 6144);
  launch_gemm(1, fused_h, 6144, op1_w_h, 6144, op1_b, nullptr, h1_h, 2048,
              B, 2048, 6144, stream);  // relu, f16-only output
  launch_gemm(0, h1_h, 2048, op2_w_h, 2048, op2_b, ln3in_f, nullptr, 1024,
              B, 1024, 2048, stream);

  // ---- 7) LN3 + learned residual mix ----
  ln3_final_kernel<<<B, 256, 0, stream>>>(ln3in_f, ln3_g, ln3_b, res_w,
                                          t_feat, s_feat, out);
}
